// VQVAE_76716705841337
// MI455X (gfx1250) — matrix operand order repaired and checked by reading, not compiled, not run on hardware
//
#include <hip/hip_runtime.h>
#include <hip/hip_bf16.h>

typedef __attribute__((ext_vector_type(16))) _Float16 v16h;
typedef __attribute__((ext_vector_type(8)))  float    v8f;

// Load a 16-half WMMA fragment as two contiguous 8-half (b128) segments.
__device__ __forceinline__ v16h ldpair(const _Float16* p0, const _Float16* p1) {
  union { uint4 u[2]; v16h h; } t;
  t.u[0] = *(const uint4*)p0;
  t.u[1] = *(const uint4*)p1;
  return t.h;
}
__device__ __forceinline__ v16h zfrag() {
  union { uint4 u[2]; v16h h; } t;
  uint4 z = {0, 0, 0, 0};
  t.u[0] = z; t.u[1] = z;
  return t.h;
}

// ---------------------------------------------------------------------------
// Weight prep -> fragment-swizzled B buffer:
//   dst[(((cls*NKT + kt)*NT + nt)*32 + lane)*16 + e]
//     = W[kh][kw][ci][co]  with k = kt*32 + (e<8 ? 0 : 16) + (lane>=16?8:0) + (e&7),
//       co = nt*16 + (lane&15), tap = k/Ci -> (ty,tx), ci = k%Ci,
//       kh = isT ? (py+pad)%stride + ty*stride : ty   (cls -> (py,px))
// ---------------------------------------------------------------------------
__global__ __launch_bounds__(256) void k_prep_wfrag(
    const float* __restrict__ src, _Float16* __restrict__ dst,
    int Co, int Ci, int KH, int KW, int stride, int pad, int isT, int srcT,
    long total) {
  long i = (long)blockIdx.x * 256 + threadIdx.x;
  if (i >= total) return;
  int e = (int)(i & 15);
  int lane = (int)((i >> 4) & 31);
  long rest = i >> 9;
  int NT = Co >> 4;
  int TY = isT ? KH / stride : KH;
  int TX = isT ? KW / stride : KW;
  int NKT = TY * TX * Ci / 32;
  int nt = (int)(rest % NT); rest /= NT;
  int kt = (int)(rest % NKT);
  int cls = (int)(rest / NKT);
  int py = isT ? cls / stride : 0, px = isT ? cls % stride : 0;
  int k = kt * 32 + ((e < 8) ? 0 : 16) + ((lane >= 16) ? 8 : 0) + (e & 7);
  int co = nt * 16 + (lane & 15);
  int tap = k / Ci, ci = k % Ci;
  int ty = tap / TX, tx = tap % TX;
  int kh = isT ? ((py + pad) % stride + ty * stride) : ty;
  int kw = isT ? ((px + pad) % stride + tx * stride) : tx;
  float v = srcT ? src[(((long)ci * Co + co) * KH + kh) * KW + kw]
                 : src[(((long)co * Ci + ci) * KH + kh) * KW + kw];
  dst[i] = (_Float16)v;
}

// codebook (512,64) f32 -> fragment-swizzled B buffer (K=64, N=512) + cnorm
//   cbf[((kt*32 + nt)*32 + lane)*16 + e] = cb[n*64 + k]
__global__ __launch_bounds__(256) void k_prep_cb(const float* __restrict__ cb,
                                                 _Float16* __restrict__ cbf,
                                                 float* __restrict__ cnorm) {
  long i = (long)blockIdx.x * 256 + threadIdx.x;
  if (i < 32768) {
    int e = (int)(i & 15);
    int lane = (int)((i >> 4) & 31);
    int rest = (int)(i >> 9);
    int nt = rest & 31, kt = rest >> 5;
    int k = kt * 32 + ((e < 8) ? 0 : 16) + ((lane >= 16) ? 8 : 0) + (e & 7);
    int n = nt * 16 + (lane & 15);
    cbf[i] = (_Float16)cb[n * 64 + k];
  }
  if (i < 512) {
    float s = 0.f;
    for (int c = 0; c < 64; ++c) {
      float v = cb[i * 64 + c];
      s += v * v;
    }
    cnorm[i] = s;
  }
}

__global__ void k_zero(float* p) {
  if (threadIdx.x == 0 && blockIdx.x == 0) p[0] = 0.f;
}

__global__ void k_final(const float* __restrict__ acc, float* __restrict__ out) {
  if (threadIdx.x == 0 && blockIdx.x == 0)
    out[6291456] = 1.25f * acc[0] / 8388608.0f;   // (1+0.25)*mean over 32*64*64*64
}

// ---------------------------------------------------------------------------
// conv1: x (32,3,256,256) NCHW f32 --k4 s2 p1 relu--> act1 (32,128,128,32) NHWC f16
// ---------------------------------------------------------------------------
__global__ __launch_bounds__(256) void k_conv1(const float* __restrict__ x,
                                               const float* __restrict__ w,
                                               const float* __restrict__ b,
                                               _Float16* __restrict__ out) {
  __shared__ float sw[32 * 3 * 16];
  __shared__ float sb[32];
  int tid = threadIdx.x;
  for (int e = tid; e < 32 * 48; e += 256) sw[e] = w[e];
  if (tid < 32) sb[tid] = b[tid];
  __syncthreads();
  long p = (long)blockIdx.x * 256 + tid;           // 32*128*128 = 524288
  int n = (int)(p >> 14);
  int rem = (int)(p & 16383);
  int oy = rem >> 7, ox = rem & 127;
  float acc[32];
#pragma unroll
  for (int co = 0; co < 32; ++co) acc[co] = sb[co];
  for (int kh = 0; kh < 4; ++kh) {
    int iy = oy * 2 - 1 + kh;
    if (iy < 0 || iy >= 256) continue;
    for (int kw = 0; kw < 4; ++kw) {
      int ix = ox * 2 - 1 + kw;
      if (ix < 0 || ix >= 256) continue;
#pragma unroll
      for (int ci = 0; ci < 3; ++ci) {
        float xv = x[(((long)n * 3 + ci) * 256 + iy) * 256 + ix];
#pragma unroll
        for (int co = 0; co < 32; ++co)
          acc[co] += xv * sw[(co * 3 + ci) * 16 + kh * 4 + kw];
      }
    }
  }
  _Float16* o = out + p * 32;
#pragma unroll
  for (int co = 0; co < 32; ++co) {
    float v = acc[co] > 0.f ? acc[co] : 0.f;
    o[co] = (_Float16)v;
  }
}

// ---------------------------------------------------------------------------
// Implicit-GEMM conv / convT via WMMA f16->f32, fully templated.
// in NHWC f16; wf fragment-swizzled weights; out NHWC f16 (+opt f32 copy).
// IS_T=1: output tiled by stride x stride parity class (blockIdx.y) so each
// 16-row M tile has a uniform 2x2 tap set -> plain GEMM per class.
// Grids are exact (M % 64 == 0, OW2 % 16 == 0): no tail handling.
// ---------------------------------------------------------------------------
template <int CIN, int COUT, int KH, int KW, int STRIDE, int IS_T, int RELU,
          int HAS32>
__global__ __launch_bounds__(128) void k_conv_wmma(
    const _Float16* __restrict__ in, int H, int W,
    const _Float16* __restrict__ wf, const float* __restrict__ bias,
    _Float16* __restrict__ out, float* __restrict__ out32, int OH, int OW,
    int NB) {
  constexpr int PAD = 1;
  constexpr int TY = IS_T ? KH / STRIDE : KH;
  constexpr int TX = IS_T ? KW / STRIDE : KW;
  constexpr int KK = TY * TX * CIN;
  constexpr int NKT = KK / 32;
  constexpr int NT = COUT / 16;
  constexpr int XS = IS_T ? STRIDE : 1;

  const int tid = threadIdx.x;
  const int wave = tid >> 5, lane = tid & 31;
  const int nb = lane & 15;

  int py = 0, px = 0;
  const _Float16* wfc = wf;
  if (IS_T) {
    const int pc = blockIdx.y;
    py = pc / STRIDE;
    px = pc % STRIDE;
    wfc = wf + (size_t)pc * NKT * NT * 512;
  }
  const int OH2 = IS_T ? OH / STRIDE : OH;
  const int OW2 = IS_T ? OW / STRIDE : OW;
  const long OHW2 = (long)OH2 * OW2;

  // Wave-uniform tile decode (tile never crosses an output row).
  const long mbase = ((long)blockIdx.x * 4 + wave) * 16;
  const int n_img = (int)(mbase / OHW2);
  const int rem = (int)(mbase % OHW2);
  const int oyb = rem / OW2, oxb = rem % OW2;

  // This lane's A-row: reduced-grid column oxb + r.
  const int r = lane & 15;
  const int kb = (lane < 16) ? 0 : 8;
  const int oxr = oxb + r;

  v8f acc[NT] = {};
  const _Float16* wl = wfc + lane * 16;

#pragma unroll
  for (int kt = 0; kt < NKT; ++kt) {
    const int ks = kt * 32;
    const int tap = ks / CIN, coff = ks % CIN;     // constant-folds post-unroll
    const int ty = tap / TX, tx = tap % TX;
    int iy, ix;
    if (IS_T) {
      iy = oyb + (py + PAD) / STRIDE - ty;         // closed form, no division
      ix = oxr + (px + PAD) / STRIDE - tx;
    } else {
      iy = oyb * STRIDE - PAD + ty;
      ix = oxr * STRIDE - PAD + tx;
    }
    v16h a = zfrag();
    if (iy >= 0 && iy < H && ix >= 0 && ix < W) {
      const _Float16* rp = in + (((long)n_img * H + iy) * W + ix) * CIN + coff;
      a = ldpair(rp + kb, rp + 16 + kb);
    }
#pragma unroll
    for (int nt = 0; nt < NT; ++nt) {
      const _Float16* bp = wl + (kt * NT + nt) * 512;
      v16h b = ldpair(bp, bp + 8);
      acc[nt] = __builtin_amdgcn_wmma_f32_16x16x32_f16(false, a, false, b,
                                                       (short)0, acc[nt],
                                                       false, false);
    }
  }

  // Store D: lane<16 -> rows v, lane>=16 -> rows v+8; col = lane&15.
  float bv[NT];
#pragma unroll
  for (int nt = 0; nt < NT; ++nt) bv[nt] = bias[nt * 16 + nb];
  const int yy = IS_T ? oyb * STRIDE + py : oyb;
  const int x0 = IS_T ? oxb * STRIDE + px : oxb;
  const long obase = (((long)n_img * OH + yy) * OW + x0) * COUT;
#pragma unroll
  for (int nt = 0; nt < NT; ++nt) {
    const int c = nt * 16 + nb;
#pragma unroll
    for (int v = 0; v < 8; ++v) {
      const int gr = (lane < 16) ? v : v + 8;
      const long o = obase + (long)gr * XS * COUT + c;
      float val = acc[nt][v] + bv[nt];
      if (RELU) val = val > 0.f ? val : 0.f;
      out[o] = (_Float16)val;
      if (HAS32) out32[o] = val;
    }
  }
}

// ---------------------------------------------------------------------------
// Vector quantizer: dist = ||c||^2 - 2 z.c  via WMMA (||z||^2 drops from
// argmin); per-row argmin via 16-lane butterfly; gather q (f32); loss accum.
// z16/z32: [131072,64]; cbf: fragment-swizzled codebook; q16: [131072,64].
// ---------------------------------------------------------------------------
__global__ __launch_bounds__(128) void k_vq(
    const _Float16* __restrict__ z16, const float* __restrict__ z32,
    const _Float16* __restrict__ cbf, const float* __restrict__ cnorm,
    const float* __restrict__ cb32, _Float16* __restrict__ q16,
    float* __restrict__ lossAcc) {
  __shared__ int sidx[64];
  __shared__ float ssum[4];
  const int tid = threadIdx.x, wave = tid >> 5, lane = tid & 31;
  const long rowbaseB = (long)blockIdx.x * 64;
  const long rowbase = rowbaseB + wave * 16;
  const int r = lane & 15;
  const int nb = lane & 15;
  const long row = rowbase + r;

  // A fragments (K=64 -> two 16x32 frags), vectorized from global.
  const int kb = (lane < 16) ? 0 : 8;
  const _Float16* zr = z16 + row * 64;
  const v16h a0 = ldpair(zr + kb, zr + 16 + kb);
  const v16h a1 = ldpair(zr + 32 + kb, zr + 48 + kb);

  float best[8];
  int bidx[8];
#pragma unroll
  for (int v = 0; v < 8; ++v) { best[v] = 3.0e38f; bidx[v] = 0; }

  const _Float16* bf = cbf + lane * 16;
  for (int nt = 0; nt < 32; ++nt) {
    const _Float16* b0p = bf + (0 * 32 + nt) * 512;
    const _Float16* b1p = bf + (1 * 32 + nt) * 512;
    v16h b0 = ldpair(b0p, b0p + 8);
    v16h b1 = ldpair(b1p, b1p + 8);
    v8f acc = {};
    acc = __builtin_amdgcn_wmma_f32_16x16x32_f16(false, a0, false, b0, (short)0,
                                                 acc, false, false);
    acc = __builtin_amdgcn_wmma_f32_16x16x32_f16(false, a1, false, b1, (short)0,
                                                 acc, false, false);
    const int ng = nt * 16 + nb;
    const float cn = cnorm[ng];
#pragma unroll
    for (int v = 0; v < 8; ++v) {
      float d = cn - 2.0f * acc[v];
      if (d < best[v]) { best[v] = d; bidx[v] = ng; }
    }
  }

  // butterfly argmin over the 16 columns in each lane half
#pragma unroll
  for (int v = 0; v < 8; ++v) {
    float bv = best[v];
    int bi = bidx[v];
    for (int off = 8; off >= 1; off >>= 1) {
      float ov = __shfl_xor(bv, off, 16);
      int oi = __shfl_xor(bi, off, 16);
      if (ov < bv) { bv = ov; bi = oi; }
    }
    if ((lane & 15) == 0) {
      int rr = (lane < 16) ? v : v + 8;
      sidx[wave * 16 + rr] = bi;
    }
  }
  __syncthreads();

  // gather q (f32 codebook), write f16 q (straight-through fwd == q), loss
  float ls = 0.f;
  for (int e = tid; e < 64 * 64; e += 128) {
    int rr = e >> 6, c = e & 63;
    long rowg = rowbaseB + rr;
    int id = sidx[rr];
    float qv = cb32[id * 64 + c];
    float zv = z32[rowg * 64 + c];
    float df = qv - zv;
    ls += df * df;
    q16[rowg * 64 + c] = (_Float16)qv;
  }
  for (int off = 16; off >= 1; off >>= 1) ls += __shfl_xor(ls, off, 32);
  if (lane == 0) ssum[wave] = ls;
  __syncthreads();
  if (tid == 0) atomicAdd(lossAcc, ssum[0] + ssum[1] + ssum[2] + ssum[3]);
}

// ---------------------------------------------------------------------------
// dec3: d2 (32,256,256,32) NHWC f16 --k3 s1 p1, sigmoid--> recon NCHW f32
// ---------------------------------------------------------------------------
__global__ __launch_bounds__(256) void k_dec3(const _Float16* __restrict__ in,
                                              const float* __restrict__ w,
                                              const float* __restrict__ b,
                                              float* __restrict__ out) {
  __shared__ float sw[3 * 32 * 9];
  __shared__ float sb[3];
  int tid = threadIdx.x;
  for (int e = tid; e < 864; e += 256) sw[e] = w[e];
  if (tid < 3) sb[tid] = b[tid];
  __syncthreads();
  long p = (long)blockIdx.x * 256 + tid;           // 32*256*256 = 2097152
  int n = (int)(p >> 16);
  int rem = (int)(p & 65535);
  int oy = rem >> 8, ox = rem & 255;
  float a0 = sb[0], a1 = sb[1], a2 = sb[2];
  for (int kh = 0; kh < 3; ++kh) {
    int iy = oy - 1 + kh;
    if (iy < 0 || iy >= 256) continue;
    for (int kw = 0; kw < 3; ++kw) {
      int ix = ox - 1 + kw;
      if (ix < 0 || ix >= 256) continue;
      const _Float16* ip = in + (((long)n * 256 + iy) * 256 + ix) * 32;
#pragma unroll
      for (int ci = 0; ci < 32; ++ci) {
        float xv = (float)ip[ci];
        a0 += xv * sw[(0 * 32 + ci) * 9 + kh * 3 + kw];
        a1 += xv * sw[(1 * 32 + ci) * 9 + kh * 3 + kw];
        a2 += xv * sw[(2 * 32 + ci) * 9 + kh * 3 + kw];
      }
    }
  }
  long base = ((long)n * 3) * 65536 + (long)oy * 256 + ox;
  out[base]          = 1.f / (1.f + __expf(-a0));
  out[base + 65536]  = 1.f / (1.f + __expf(-a1));
  out[base + 131072] = 1.f / (1.f + __expf(-a2));
}

// ---------------------------------------------------------------------------
extern "C" void kernel_launch(void* const* d_in, const int* in_sizes, int n_in,
                              void* d_out, int out_size, void* d_ws,
                              size_t ws_size, hipStream_t stream) {
  (void)in_sizes; (void)n_in; (void)out_size; (void)ws_size;
  const float* x        = (const float*)d_in[0];
  const float* enc_w1   = (const float*)d_in[1];
  const float* enc_b1   = (const float*)d_in[2];
  const float* enc_w2   = (const float*)d_in[3];
  const float* enc_b2   = (const float*)d_in[4];
  const float* enc_w3   = (const float*)d_in[5];
  const float* enc_b3   = (const float*)d_in[6];
  const float* codebook = (const float*)d_in[7];
  const float* dec_w1   = (const float*)d_in[8];
  const float* dec_b1   = (const float*)d_in[9];
  const float* dec_w2   = (const float*)d_in[10];
  const float* dec_b2   = (const float*)d_in[11];
  const float* dec_w3   = (const float*)d_in[12];
  const float* dec_b3   = (const float*)d_in[13];
  float* out = (float*)d_out;
  char* ws = (char*)d_ws;

  // Workspace layout with buffer reuse (~219 MB):
  //  P0 [134 MB]: act1|act2|z16 during encode, then d2 during decode
  //  P1 [ 67 MB]: z32 during encode/VQ, then d1 during decode
  //  P2 [ 16 MB]: q16
  //  P3        : fragment-swizzled weights, codebook, scalars
  const size_t P0 = 0;
  const size_t P1 = P0 + (size_t)134217728;
  const size_t P2 = P1 + (size_t)67108864;
  const size_t P3 = P2 + (size_t)16777216;
  _Float16* act1 = (_Float16*)(ws + P0);
  _Float16* act2 = (_Float16*)(ws + P0 + 33554432);
  _Float16* z16  = (_Float16*)(ws + P0 + 50331648);
  _Float16* d2   = (_Float16*)(ws + P0);
  float*    z32  = (float*)(ws + P1);
  _Float16* d1   = (_Float16*)(ws + P1);
  _Float16* q16  = (_Float16*)(ws + P2);
  size_t mo = P3;
  _Float16* w2f  = (_Float16*)(ws + mo); mo += (size_t)16 * 4 * 512 * 2;      // 64KB
  _Float16* w3f  = (_Float16*)(ws + mo); mo += (size_t)18 * 4 * 512 * 2;      // 72KB
  _Float16* wd1f = (_Float16*)(ws + mo); mo += (size_t)4 * 8 * 4 * 512 * 2;   // 128KB
  _Float16* wd2f = (_Float16*)(ws + mo); mo += (size_t)4 * 8 * 2 * 512 * 2;   // 64KB
  _Float16* cbf  = (_Float16*)(ws + mo); mo += (size_t)2 * 32 * 512 * 2;      // 64KB
  float* cnorm   = (float*)(ws + mo);    mo += (size_t)512 * 4;
  float* lossAcc = (float*)(ws + mo);    mo += 256;

  // ---- weight / codebook prep (fragment-swizzled B buffers)
  k_prep_wfrag<<<128, 256, 0, stream>>>(enc_w2, w2f, 64, 32, 4, 4, 2, 1, 0, 0,
                                        (long)16 * 4 * 512);
  k_prep_wfrag<<<144, 256, 0, stream>>>(enc_w3, w3f, 64, 64, 3, 3, 1, 1, 0, 0,
                                        (long)18 * 4 * 512);
  k_prep_wfrag<<<256, 256, 0, stream>>>(dec_w1, wd1f, 64, 64, 4, 4, 2, 1, 1, 1,
                                        (long)4 * 8 * 4 * 512);
  k_prep_wfrag<<<128, 256, 0, stream>>>(dec_w2, wd2f, 32, 64, 4, 4, 2, 1, 1, 1,
                                        (long)4 * 8 * 2 * 512);
  k_prep_cb<<<128, 256, 0, stream>>>(codebook, cbf, cnorm);
  k_zero<<<1, 64, 0, stream>>>(lossAcc);

  // ---- encoder
  k_conv1<<<2048, 256, 0, stream>>>(x, enc_w1, enc_b1, act1);
  // conv2: 128x128x32 -> 64x64x64, k4 s2 p1, relu   (M=131072)
  k_conv_wmma<32, 64, 4, 4, 2, 0, 1, 0>
      <<<dim3(2048, 1), 128, 0, stream>>>(act1, 128, 128, w2f, enc_b2, act2,
                                          nullptr, 64, 64, 32);
  // conv3: 64x64x64 -> 64x64x64, k3 s1 p1           (z in f16 + f32)
  k_conv_wmma<64, 64, 3, 3, 1, 0, 0, 1>
      <<<dim3(2048, 1), 128, 0, stream>>>(act2, 64, 64, w3f, enc_b3, z16, z32,
                                          64, 64, 32);
  // ---- vector quantizer + loss
  k_vq<<<2048, 128, 0, stream>>>(z16, z32, cbf, cnorm, codebook, q16, lossAcc);
  k_final<<<1, 64, 0, stream>>>(lossAcc, out);

  // ---- decoder
  // convT1: 64x64x64 -> 128x128x64, k4 s2 p1, relu  (M/class = 131072)
  k_conv_wmma<64, 64, 4, 4, 2, 1, 1, 0>
      <<<dim3(2048, 4), 128, 0, stream>>>(q16, 64, 64, wd1f, dec_b1, d1,
                                          nullptr, 128, 128, 32);
  // convT2: 128x128x64 -> 256x256x32, k4 s2 p1, relu (M/class = 524288)
  k_conv_wmma<64, 32, 4, 4, 2, 1, 1, 0>
      <<<dim3(8192, 4), 128, 0, stream>>>(d1, 128, 128, wd2f, dec_b2, d2,
                                          nullptr, 256, 256, 32);
  // dec3 + sigmoid -> recon NCHW f32
  k_dec3<<<8192, 256, 0, stream>>>(d2, dec_w3, dec_b3, out);
}
